// TreeModel_33663953666586
// MI455X (gfx1250) — matrix operand order repaired and checked by reading, not compiled, not run on hardware
//
#include <hip/hip_runtime.h>

// ---------------------------------------------------------------------------
// Fused soft-decision-tree forward for MI455X (gfx1250, wave32, WMMA bf16)
//   gates = sigmoid(x @ W^T + b)        GEMM1: 16384x1024 @ 1024x511
//   prob  = tree expansion (9 levels)   per-row path products
//   out   = prob @ leaf_logits          GEMM2: 16384x512 @ 512x100
// Round 2: dual independent WMMA accumulator chains (even/odd K-steps) to
// break the serial D->C dependency — required ILP since 297KB LDS leaves
// exactly 1 wave per SIMD32 (no cross-wave latency hiding).
// ---------------------------------------------------------------------------

typedef __bf16 v16bf __attribute__((ext_vector_type(16)));
typedef float  v8f   __attribute__((ext_vector_type(8)));

#define BLOCK_M  64
#define THREADS  128           // 4 waves (wave32)
#define FEAT     1024
#define N_INNER  511
#define NCLS     100
#define NLEAF    512
#define BATCH    16384

// LDS layout (bytes). Rows padded (+8 bf16 / +4 f32) to stagger banks.
#define XSTR   1032            // bf16 elements per x-stage row
#define WSTR   1032            // bf16 elements per W-tile row
#define GSTR   516             // f32  elements per gates row
#define LPSTR  520             // bf16 elements per leaf-prob row
#define LSSTR  520             // bf16 elements per leaf-logit^T row
#define OFF_X   0
#define OFF_W   (BLOCK_M * XSTR * 2)            // 132096
#define OFF_G   (OFF_W + 16 * WSTR * 2)         // 165120
#define OFF_LP  0                               // overlays dead x-stage
#define OFF_LS  OFF_G                           // overlays dead gates
#define SMEM_BYTES (OFF_G + BLOCK_M * GSTR * 4) // 297216 (< 320 KB WGP LDS)

static __device__ __forceinline__ unsigned pack2bf(float a, float b) {
    union { __bf16 h[2]; unsigned u; } p;
    p.h[0] = (__bf16)a;
    p.h[1] = (__bf16)b;
    return p.u;
}

__global__ void __launch_bounds__(THREADS)
tree_fused_kernel(const float* __restrict__ x, const float* __restrict__ W,
                  const float* __restrict__ b, const float* __restrict__ leaf,
                  float* __restrict__ out)
{
    extern __shared__ __align__(16) char smem[];
    __bf16* XS = (__bf16*)(smem + OFF_X);   // x tile, bf16 [64][XSTR]
    __bf16* WT = (__bf16*)(smem + OFF_W);   // W n-tile, bf16 [16][WSTR]
    float*  G  = (float* )(smem + OFF_G);   // gates f32 [64][GSTR]
    __bf16* LP = (__bf16*)(smem + OFF_LP);  // leaf probs bf16 [64][LPSTR]
    __bf16* LS = (__bf16*)(smem + OFF_LS);  // leaf_logits^T bf16 [112][LSSTR]

    const int tid  = threadIdx.x;
    const int lane = tid & 31;
    const int wave = tid >> 5;
    const int blk  = blockIdx.x;

    const int nloc = lane & 15;          // N (and A-row M) within 16
    const int hi   = (lane >> 4) & 1;    // lane-half selector
    const int sA   = hi ? 16 : 0;        // A K-half (contiguous, matches B)
    const int sB   = hi ? 16 : 0;        // B K-half (contiguous, matches A)
    const int rowA = wave * 16 + nloc;   // local row owned by this lane for A

    // ---- stage x tile: 64x1024 f32 -> bf16 in LDS (rows are contiguous) ----
    {
        const float4* src = (const float4*)(x + (size_t)blk * (BLOCK_M * FEAT));
        for (int v = tid; v < BLOCK_M * FEAT / 4; v += THREADS) {
            float4 f = src[v];
            int e = v * 4;
            int r = e >> 10, c = e & 1023;
            uint2 p;
            p.x = pack2bf(f.x, f.y);
            p.y = pack2bf(f.z, f.w);
            *(uint2*)(XS + r * XSTR + c) = p;
        }
    }
    __syncthreads();

    const __bf16* aBase1 = XS + rowA * XSTR + sA;
    const __bf16* bBase1 = WT + nloc * WSTR + sB;

    // =========================== GEMM1 + sigmoid ===========================
    for (int t = 0; t < 32; ++t) {
        __syncthreads();                      // previous W tile fully consumed
        // stage W n-tile: rows 16t..16t+15 (guard row 511), f32 -> bf16
        for (int v = tid; v < 16 * FEAT / 4; v += THREADS) {
            int e = v * 4;
            int gidx = t * (16 * FEAT) + e;   // W rows are contiguous
            float4 f = make_float4(0.f, 0.f, 0.f, 0.f);
            if (gidx < N_INNER * FEAT) f = *(const float4*)(W + gidx);
            int r = e >> 10, c = e & 1023;
            uint2 p;
            p.x = pack2bf(f.x, f.y);
            p.y = pack2bf(f.z, f.w);
            *(uint2*)(WT + r * WSTR + c) = p;
        }
        if (t < 31)  // pull next tile toward L2/L0 (global_prefetch_b8)
            __builtin_prefetch(W + (size_t)(t + 1) * (16 * FEAT) + tid * 16, 0, 0);
        __syncthreads();

        // Two independent accumulation chains (even/odd K-steps).
        v8f acc0 = {0.f, 0.f, 0.f, 0.f, 0.f, 0.f, 0.f, 0.f};
        v8f acc1 = {0.f, 0.f, 0.f, 0.f, 0.f, 0.f, 0.f, 0.f};
        #pragma unroll 4
        for (int ks = 0; ks < 32; ks += 2) {
            union { v16bf v; uint4 q[2]; } A0, B0, A1, B1;
            A0.q[0] = *(const uint4*)(aBase1 + ks * 32);
            A0.q[1] = *(const uint4*)(aBase1 + ks * 32 + 8);
            B0.q[0] = *(const uint4*)(bBase1 + ks * 32);
            B0.q[1] = *(const uint4*)(bBase1 + ks * 32 + 8);
            A1.q[0] = *(const uint4*)(aBase1 + ks * 32 + 32);
            A1.q[1] = *(const uint4*)(aBase1 + ks * 32 + 40);
            B1.q[0] = *(const uint4*)(bBase1 + ks * 32 + 32);
            B1.q[1] = *(const uint4*)(bBase1 + ks * 32 + 40);
            acc0 = __builtin_amdgcn_wmma_f32_16x16x32_bf16(
                       false, A0.v, false, B0.v, (short)0, acc0, false, false);
            acc1 = __builtin_amdgcn_wmma_f32_16x16x32_bf16(
                       false, A1.v, false, B1.v, (short)0, acc1, false, false);
        }
        v8f acc = acc0 + acc1;

        // bias + sigmoid; scatter per C/D layout: N=lane%16, M=r+8*hi
        int col = t * 16 + nloc;
        float bias = (col < N_INNER) ? b[col] : 0.f;
        #pragma unroll
        for (int r = 0; r < 8; ++r) {
            int m = r + (hi ? 8 : 0);
            float lg = acc[r] + bias;
            float g  = 1.0f / (1.0f + __expf(-lg));
            G[(wave * 16 + m) * GSTR + col] = g;
        }
    }
    __syncthreads();   // gates complete; x-stage dead

    // ======================= tree expansion (9 levels) =====================
    // task = (row, 16-leaf-group-of-32): shared 4-level prefix, then
    // fully unrolled in-register 5-level expansion (31 gate reads, 62 mults).
    for (int task = tid; task < BLOCK_M * 16; task += THREADS) {
        int row = task >> 4;
        int grp = task & 15;
        const float* g = G + row * GSTR;

        float p = 1.f;
        #pragma unroll
        for (int l = 0; l < 4; ++l) {
            int node = (1 << l) - 1 + (grp >> (4 - l));
            float gv = g[node];
            p *= ((grp >> (3 - l)) & 1) ? gv : (1.f - gv);
        }

        float pr[32];
        pr[0] = p;
        #pragma unroll
        for (int l = 4; l <= 8; ++l) {
            int cnt  = 1 << (l - 4);
            int base = (1 << l) - 1 + (grp << (l - 4));
            #pragma unroll
            for (int i = cnt - 1; i >= 0; --i) {
                float gv = g[base + i];
                float pi = pr[i];
                pr[2 * i + 1] = pi * gv;           // right child: * p
                pr[2 * i]     = pi * (1.f - gv);   // left child:  * (1-p)
            }
        }
        unsigned* dst = (unsigned*)(LP + row * LPSTR + grp * 32);
        #pragma unroll
        for (int i = 0; i < 16; ++i)
            dst[i] = pack2bf(pr[2 * i], pr[2 * i + 1]);
    }
    __syncthreads();   // all gate reads done; gates region dead

    // ---- stage leaf_logits^T: [512][100] f32 -> [112][512] bf16 ----
    for (int idx = tid; idx < 112 * NLEAF; idx += THREADS) {
        int j = idx / 112;
        int c = idx - j * 112;
        float v = (c < NCLS) ? leaf[j * NCLS + c] : 0.f;
        LS[c * LSSTR + j] = (__bf16)v;
    }
    __syncthreads();

    // ================================ GEMM2 ================================
    const __bf16* aBase2 = LP + rowA * LPSTR + sA;
    for (int t2 = 0; t2 < 7; ++t2) {
        const __bf16* bBase2 = LS + (t2 * 16 + nloc) * LSSTR + sB;
        v8f acc0 = {0.f, 0.f, 0.f, 0.f, 0.f, 0.f, 0.f, 0.f};
        v8f acc1 = {0.f, 0.f, 0.f, 0.f, 0.f, 0.f, 0.f, 0.f};
        #pragma unroll
        for (int ks = 0; ks < 16; ks += 2) {
            union { v16bf v; uint4 q[2]; } A0, B0, A1, B1;
            A0.q[0] = *(const uint4*)(aBase2 + ks * 32);
            A0.q[1] = *(const uint4*)(aBase2 + ks * 32 + 8);
            B0.q[0] = *(const uint4*)(bBase2 + ks * 32);
            B0.q[1] = *(const uint4*)(bBase2 + ks * 32 + 8);
            A1.q[0] = *(const uint4*)(aBase2 + ks * 32 + 32);
            A1.q[1] = *(const uint4*)(aBase2 + ks * 32 + 40);
            B1.q[0] = *(const uint4*)(bBase2 + ks * 32 + 32);
            B1.q[1] = *(const uint4*)(bBase2 + ks * 32 + 40);
            acc0 = __builtin_amdgcn_wmma_f32_16x16x32_bf16(
                       false, A0.v, false, B0.v, (short)0, acc0, false, false);
            acc1 = __builtin_amdgcn_wmma_f32_16x16x32_bf16(
                       false, A1.v, false, B1.v, (short)0, acc1, false, false);
        }
        v8f acc = acc0 + acc1;

        int col = t2 * 16 + nloc;
        if (col < NCLS) {
            #pragma unroll
            for (int r = 0; r < 8; ++r) {
                int m = r + (hi ? 8 : 0);
                out[((size_t)blk * BLOCK_M + wave * 16 + m) * NCLS + col] = acc[r];
            }
        }
    }
}

extern "C" void kernel_launch(void* const* d_in, const int* in_sizes, int n_in,
                              void* d_out, int out_size, void* d_ws, size_t ws_size,
                              hipStream_t stream) {
    (void)in_sizes; (void)n_in; (void)out_size; (void)d_ws; (void)ws_size;
    const float* x    = (const float*)d_in[0];
    const float* W    = (const float*)d_in[1];
    const float* b    = (const float*)d_in[2];
    const float* leaf = (const float*)d_in[3];
    float* out = (float*)d_out;

    hipFuncSetAttribute((const void*)tree_fused_kernel,
                        hipFuncAttributeMaxDynamicSharedMemorySize, SMEM_BYTES);
    tree_fused_kernel<<<BATCH / BLOCK_M, THREADS, SMEM_BYTES, stream>>>(
        x, W, b, leaf, out);
}
